// RelativeSelfAttentionLayer_60662118088866
// MI455X (gfx1250) — compile-verified, matrix operand order checked by reading, not run on hardware
//
#include <hip/hip_runtime.h>

#define TT 1024
#define CC 512
#define HH 8
#define DD 64
#define BB 4
#define PPAD 2048   // 2T-1 = 2047 padded to 2048
#define ICHUNK 128  // pos-score row chunk

typedef __attribute__((ext_vector_type(16))) __bf16 v16bf;
typedef __attribute__((ext_vector_type(8)))  float  v8f;

union FragBF {
  v16bf bf;
  uint4 u4[2];
  unsigned short us[16];
};

__device__ __forceinline__ unsigned short f2bf(float f) {
  unsigned int u = __float_as_uint(f);
  u += 0x7FFFu + ((u >> 16) & 1u);   // round-to-nearest-even
  return (unsigned short)(u >> 16);
}

__device__ __forceinline__ void store_bf8(unsigned short* dst, const float* v) {
  uint4 pk;
  pk.x = (unsigned)f2bf(v[0]) | ((unsigned)f2bf(v[1]) << 16);
  pk.y = (unsigned)f2bf(v[2]) | ((unsigned)f2bf(v[3]) << 16);
  pk.z = (unsigned)f2bf(v[4]) | ((unsigned)f2bf(v[5]) << 16);
  pk.w = (unsigned)f2bf(v[6]) | ((unsigned)f2bf(v[7]) << 16);
  *reinterpret_cast<uint4*>(dst) = pk;
}

__device__ __forceinline__ v8f wmma_bf16(const FragBF& a, const FragBF& b, v8f c) {
  // D = A(16x32 bf16) x B(32x16 bf16) + C(16x16 f32)
  return __builtin_amdgcn_wmma_f32_16x16x32_bf16(false, a.bf, false, b.bf,
                                                 (short)0, c, false, false);
}

__device__ __forceinline__ v8f vzero8() {
  v8f z = {0.f, 0.f, 0.f, 0.f, 0.f, 0.f, 0.f, 0.f};
  return z;
}

// ---------------------------------------------------------------------------
// Weights -> bf16 (row-major kept: rows are WMMA A-fragment M, K contiguous),
// plus a 512-entry fp32 zero-bias vector so GEMM epilogues never branch.
__global__ void convert_w_kernel(const float* __restrict__ Wq, const float* __restrict__ Wk,
                                 const float* __restrict__ Wv, const float* __restrict__ Wp,
                                 const float* __restrict__ Wo, unsigned short* __restrict__ dst,
                                 float* __restrict__ zbias) {
  int idx = blockIdx.x * blockDim.x + threadIdx.x;        // 5 * 2^18 + 512
  const int total = 5 * 262144;
  if (idx < total) {
    int which = idx >> 18;
    int off = idx & 262143;
    const float* src = (which == 0) ? Wq : (which == 1) ? Wk : (which == 2) ? Wv
                      : (which == 3) ? Wp : Wo;
    dst[idx] = f2bf(src[off]);
  } else if (idx < total + CC) {
    zbias[idx - total] = 0.f;
  }
}

// pos_emb [1,C,2T-1] -> posT bf16 [PPAD, C] (row j = one position, K=c contiguous)
__global__ void transpose_pos_kernel(const float* __restrict__ pos, unsigned short* __restrict__ posT) {
  int idx = blockIdx.x * blockDim.x + threadIdx.x;        // PPAD*CC
  if (idx >= PPAD * CC) return;
  int j = idx / CC, c = idx - j * CC;
  posT[idx] = (j < 2 * TT - 1) ? f2bf(pos[(size_t)c * (2 * TT - 1) + j]) : (unsigned short)0;
}

// Channels-first LayerNorm over C at each (b,t); writes xnT bf16 [N=B*T, C]
__global__ void layernorm_kernel(const float* __restrict__ x, const float* __restrict__ g,
                                 const float* __restrict__ be, unsigned short* __restrict__ xnT) {
  const int n = blockIdx.x;              // 0..4095
  const int b = n >> 10, t = n & (TT - 1);
  const int tid = threadIdx.x;           // 128 threads
  float v[4];
  float s = 0.f, sq = 0.f;
  #pragma unroll
  for (int k = 0; k < 4; ++k) {
    int c = tid + 128 * k;
    float xv = x[((size_t)b * CC + c) * TT + t];
    v[k] = xv; s += xv; sq += xv * xv;
  }
  __shared__ float r0[128], r1[128];
  r0[tid] = s; r1[tid] = sq; __syncthreads();
  for (int off = 64; off > 0; off >>= 1) {
    if (tid < off) { r0[tid] += r0[tid + off]; r1[tid] += r1[tid + off]; }
    __syncthreads();
  }
  const float mean = r0[0] * (1.f / CC);
  const float var  = r1[0] * (1.f / CC) - mean * mean;
  const float rstd = rsqrtf(var + 1e-5f);
  #pragma unroll
  for (int k = 0; k < 4; ++k) {
    int c = tid + 128 * k;
    xnT[(size_t)n * CC + c] = f2bf((v[k] - mean) * rstd * g[c] + be[c]);
  }
}

// ---------------------------------------------------------------------------
// Generic 512-K WMMA GEMM: Y[o,n] = sum_c W[o,c] * X[n,c]  (X stored [n][c])
// Block = 4 waves (64 M rows); each wave computes a 16x64 strip:
// one A fragment is reused across 4 B fragments -> 4 WMMAs per K-step.
//  mode 0: Q  -> qu,qv bf16 [B,H,T,D] (+bq, +u_bias / +v_bias)
//  mode 1: K  -> kT    bf16 [B,H,T,D] (+bk)
//  mode 2: V  -> v     bf16 [B,H,D,T] (+bv)
//  mode 3: P  -> pT    bf16 [H,PPAD,D] (zero bias)
//  mode 4: O  -> out   f32  [B,C,T]   (+bo)
__global__ void gemm512_kernel(const unsigned short* __restrict__ W,
                               const unsigned short* __restrict__ X,
                               const float* __restrict__ bias,
                               const float* __restrict__ ub,
                               const float* __restrict__ vb,
                               void* __restrict__ out0, void* __restrict__ out1,
                               int mode) {
  const int lane = threadIdx.x & 31, wid = threadIdx.x >> 5;
  const int l2 = lane & 15, hh = lane >> 4;
  const int m0 = (blockIdx.x * 4 + wid) * 16;
  const int n0 = blockIdx.y * 64;
  const unsigned short* Arow = W + (size_t)(m0 + l2) * CC;
  const unsigned short* Brow[4];
  #pragma unroll
  for (int j = 0; j < 4; ++j) Brow[j] = X + (size_t)(n0 + 16 * j + l2) * CC;

  v8f acc[4];
  #pragma unroll
  for (int j = 0; j < 4; ++j) acc[j] = vzero8();

  #pragma unroll 2
  for (int k0 = 0; k0 < CC; k0 += 32) {
    __builtin_prefetch(Arow + k0 + 128, 0, 1);
    FragBF a;
    a.u4[0] = *reinterpret_cast<const uint4*>(Arow + k0 + hh * 8);
    a.u4[1] = *reinterpret_cast<const uint4*>(Arow + k0 + 16 + hh * 8);
    #pragma unroll
    for (int j = 0; j < 4; ++j) {
      FragBF b;
      b.u4[0] = *reinterpret_cast<const uint4*>(Brow[j] + k0 + hh * 8);
      b.u4[1] = *reinterpret_cast<const uint4*>(Brow[j] + k0 + 16 + hh * 8);
      acc[j] = wmma_bf16(a, b, acc[j]);
    }
  }

  const int mbase = m0 + 8 * hh;          // C/D layout: VGPR r -> M = r (+8 for hi lanes)
  float yb[8];
  #pragma unroll
  for (int r = 0; r < 8; ++r) yb[r] = bias[mbase + r];

  #pragma unroll
  for (int j = 0; j < 4; ++j) {
    const int n = n0 + 16 * j + l2;
    float y[8];
    #pragma unroll
    for (int r = 0; r < 8; ++r) y[r] = acc[j][r] + yb[r];

    if (mode == 0) {
      const int b = n >> 10, t = n & (TT - 1);
      const int h = mbase >> 6, d0 = mbase & (DD - 1);
      float yu[8], yv[8];
      #pragma unroll
      for (int r = 0; r < 8; ++r) { yu[r] = y[r] + ub[mbase + r]; yv[r] = y[r] + vb[mbase + r]; }
      size_t base = ((size_t)(b * HH + h) * TT + t) * DD + d0;
      store_bf8((unsigned short*)out0 + base, yu);
      store_bf8((unsigned short*)out1 + base, yv);
    } else if (mode == 1) {
      const int b = n >> 10, t = n & (TT - 1);
      const int h = mbase >> 6, d0 = mbase & (DD - 1);
      store_bf8((unsigned short*)out0 + ((size_t)(b * HH + h) * TT + t) * DD + d0, y);
    } else if (mode == 2) {
      const int b = n >> 10, t = n & (TT - 1);
      unsigned short* dst = (unsigned short*)out0;
      #pragma unroll
      for (int r = 0; r < 8; ++r) {
        int m = mbase + r, h = m >> 6, d = m & (DD - 1);
        dst[((size_t)(b * HH + h) * DD + d) * TT + t] = f2bf(y[r]);
      }
    } else if (mode == 3) {
      const int h = mbase >> 6, d0 = mbase & (DD - 1);
      store_bf8((unsigned short*)out0 + ((size_t)h * PPAD + n) * DD + d0, y);
    } else {
      const int b = n >> 10, t = n & (TT - 1);
      float* dst = (float*)out0;
      #pragma unroll
      for (int r = 0; r < 8; ++r)
        dst[((size_t)b * CC + (mbase + r)) * TT + t] = y[r];
    }
  }
}

// content_score[bh,i,j] = sum_d qu[bh,i,d] * kT[bh,j,d]   (K=64)
// Block = 4 waves x 16 rows; wave strip = 16x64 (A reused over 4 B frags).
__global__ void content_gemm_kernel(const unsigned short* __restrict__ qu,
                                    const unsigned short* __restrict__ kT,
                                    float* __restrict__ score) {
  const int lane = threadIdx.x & 31, wid = threadIdx.x >> 5;
  const int l2 = lane & 15, hh = lane >> 4;
  const int i0 = (blockIdx.x * 4 + wid) * 16;
  const int j0 = blockIdx.y * 64;
  const int bh = blockIdx.z;
  const unsigned short* A = qu + ((size_t)bh * TT + i0 + l2) * DD;
  const unsigned short* Bp[4];
  #pragma unroll
  for (int j = 0; j < 4; ++j) Bp[j] = kT + ((size_t)bh * TT + j0 + 16 * j + l2) * DD;

  v8f acc[4];
  #pragma unroll
  for (int j = 0; j < 4; ++j) acc[j] = vzero8();

  #pragma unroll
  for (int k0 = 0; k0 < DD; k0 += 32) {
    FragBF a;
    a.u4[0] = *reinterpret_cast<const uint4*>(A + k0 + hh * 8);
    a.u4[1] = *reinterpret_cast<const uint4*>(A + k0 + 16 + hh * 8);
    #pragma unroll
    for (int j = 0; j < 4; ++j) {
      FragBF b;
      b.u4[0] = *reinterpret_cast<const uint4*>(Bp[j] + k0 + hh * 8);
      b.u4[1] = *reinterpret_cast<const uint4*>(Bp[j] + k0 + 16 + hh * 8);
      acc[j] = wmma_bf16(a, b, acc[j]);
    }
  }
  float* sbase = score + (size_t)bh * TT * TT;
  #pragma unroll
  for (int j = 0; j < 4; ++j) {
    const int jj = j0 + 16 * j + l2;
    #pragma unroll
    for (int r = 0; r < 8; ++r)
      sbase[(size_t)(i0 + 8 * hh + r) * TT + jj] = acc[j][r];
  }
}

// raw[bh, ii, jj] = sum_d qv[bh, i0c+ii, d] * pT[h, jj, d]   for one 128-row chunk
__global__ void pos_gemm_kernel(const unsigned short* __restrict__ qv,
                                const unsigned short* __restrict__ pT,
                                float* __restrict__ raw, int i0c) {
  const int lane = threadIdx.x & 31, wid = threadIdx.x >> 5;
  const int l2 = lane & 15, hh = lane >> 4;
  const int i0 = (blockIdx.x * 4 + wid) * 16;     // within chunk
  const int j0 = blockIdx.y * 64;
  const int bh = blockIdx.z;
  const int h = bh & (HH - 1);
  const unsigned short* A = qv + ((size_t)bh * TT + i0c + i0 + l2) * DD;
  const unsigned short* Bp[4];
  #pragma unroll
  for (int j = 0; j < 4; ++j) Bp[j] = pT + ((size_t)h * PPAD + j0 + 16 * j + l2) * DD;

  v8f acc[4];
  #pragma unroll
  for (int j = 0; j < 4; ++j) acc[j] = vzero8();

  #pragma unroll
  for (int k0 = 0; k0 < DD; k0 += 32) {
    FragBF a;
    a.u4[0] = *reinterpret_cast<const uint4*>(A + k0 + hh * 8);
    a.u4[1] = *reinterpret_cast<const uint4*>(A + k0 + 16 + hh * 8);
    #pragma unroll
    for (int j = 0; j < 4; ++j) {
      FragBF b;
      b.u4[0] = *reinterpret_cast<const uint4*>(Bp[j] + k0 + hh * 8);
      b.u4[1] = *reinterpret_cast<const uint4*>(Bp[j] + k0 + 16 + hh * 8);
      acc[j] = wmma_bf16(a, b, acc[j]);
    }
  }
  float* rbase = raw + (size_t)bh * ICHUNK * PPAD;
  #pragma unroll
  for (int j = 0; j < 4; ++j) {
    const int jj = j0 + 16 * j + l2;
    #pragma unroll
    for (int r = 0; r < 8; ++r)
      rbase[(size_t)(i0 + 8 * hh + r) * PPAD + jj] = acc[j][r];
  }
}

// rel_shift fused add: score[bh,i,j] += raw[bh, i-i0c, j + T-1 - i]
__global__ void shift_add_kernel(float* __restrict__ score, const float* __restrict__ raw, int i0c) {
  size_t idx = (size_t)blockIdx.x * blockDim.x + threadIdx.x;  // 32*128*1024
  if (idx >= (size_t)BB * HH * ICHUNK * TT) return;
  int j = (int)(idx & (TT - 1));
  int ii = (int)((idx >> 10) & (ICHUNK - 1));
  int bh = (int)(idx >> 17);
  int i = i0c + ii;
  score[((size_t)bh * TT + i) * TT + j] +=
      raw[((size_t)bh * ICHUNK + ii) * PPAD + (j + TT - 1 - i)];
}

// Scale + mask + softmax over j for row (bh,i); write bf16 TRANSPOSED: attnT[bh, j, i]
// (so the context GEMM's B-fragment columns are K-contiguous).
__global__ void softmax_kernel(const float* __restrict__ score, const int* __restrict__ x_mask,
                               unsigned short* __restrict__ attnT) {
  const int row = blockIdx.x;           // bh*T + i
  const int bh = row >> 10, i = row & (TT - 1), b = bh >> 3;
  const float* s = score + (size_t)row * TT;
  const int mi = x_mask[b * TT + i];
  const float scale = 0.044194173824159216f;   // 1/sqrt(512)
  __shared__ float red[256];
  const int tid = threadIdx.x;
  float vals[4];
  float mx = -3.4e38f;
  #pragma unroll
  for (int k = 0; k < 4; ++k) {
    int j = tid + 256 * k;
    float v = s[j] * scale;
    if (mi == 0 || x_mask[b * TT + j] == 0) v = -10000.f;
    vals[k] = v; mx = fmaxf(mx, v);
  }
  red[tid] = mx; __syncthreads();
  for (int off = 128; off > 0; off >>= 1) {
    if (tid < off) red[tid] = fmaxf(red[tid], red[tid + off]);
    __syncthreads();
  }
  mx = red[0]; __syncthreads();
  float sum = 0.f;
  #pragma unroll
  for (int k = 0; k < 4; ++k) { vals[k] = __expf(vals[k] - mx); sum += vals[k]; }
  red[tid] = sum; __syncthreads();
  for (int off = 128; off > 0; off >>= 1) {
    if (tid < off) red[tid] += red[tid + off];
    __syncthreads();
  }
  const float inv = 1.f / red[0];
  #pragma unroll
  for (int k = 0; k < 4; ++k) {
    int j = tid + 256 * k;
    attnT[((size_t)bh * TT + j) * TT + i] = f2bf(vals[k] * inv);
  }
}

// context[bh,d,t] = sum_s v[bh,d,s] * attn[s,t]; attn read via attnT[t][s].
// Block = 4 waves; wave `wid` owns d-tile wid*16; wave strip = 16x64 over t.
// Output written channels-last: ctxT[(b*T+t)*C + h*64 + d] (bf16).
__global__ void context_gemm_kernel(const unsigned short* __restrict__ vbf,
                                    const unsigned short* __restrict__ attnT,
                                    unsigned short* __restrict__ ctxT) {
  const int lane = threadIdx.x & 31, wid = threadIdx.x >> 5;
  const int l2 = lane & 15, hh = lane >> 4;
  const int t0 = blockIdx.x * 64;
  const int bh = blockIdx.y;
  const int b = bh >> 3, h = bh & (HH - 1);
  const unsigned short* A = vbf + ((size_t)bh * DD + wid * 16 + l2) * TT;
  const unsigned short* Bp[4];
  #pragma unroll
  for (int j = 0; j < 4; ++j) Bp[j] = attnT + ((size_t)bh * TT + t0 + 16 * j + l2) * TT;

  v8f acc[4];
  #pragma unroll
  for (int j = 0; j < 4; ++j) acc[j] = vzero8();

  #pragma unroll 2
  for (int k0 = 0; k0 < TT; k0 += 32) {
    __builtin_prefetch(A + k0 + 128, 0, 1);
    FragBF a;
    a.u4[0] = *reinterpret_cast<const uint4*>(A + k0 + hh * 8);
    a.u4[1] = *reinterpret_cast<const uint4*>(A + k0 + 16 + hh * 8);
    #pragma unroll
    for (int j = 0; j < 4; ++j) {
      FragBF bf2;
      bf2.u4[0] = *reinterpret_cast<const uint4*>(Bp[j] + k0 + hh * 8);
      bf2.u4[1] = *reinterpret_cast<const uint4*>(Bp[j] + k0 + 16 + hh * 8);
      acc[j] = wmma_bf16(a, bf2, acc[j]);
    }
  }
  const int d0 = wid * 16 + 8 * hh;
  #pragma unroll
  for (int j = 0; j < 4; ++j) {
    const int t = t0 + 16 * j + l2;
    float y[8];
    #pragma unroll
    for (int r = 0; r < 8; ++r) y[r] = acc[j][r];
    store_bf8(ctxT + ((size_t)(b * TT + t) * CC + h * DD + d0), y);
  }
}

// ---------------------------------------------------------------------------
extern "C" void kernel_launch(void* const* d_in, const int* in_sizes, int n_in,
                              void* d_out, int out_size, void* d_ws, size_t ws_size,
                              hipStream_t stream) {
  const float* x       = (const float*)d_in[0];
  const float* pos_emb = (const float*)d_in[1];
  const int*   x_mask  = (const int*)d_in[2];
  const float* ln_g    = (const float*)d_in[3];
  const float* ln_b    = (const float*)d_in[4];
  const float* Wq = (const float*)d_in[5];
  const float* bq = (const float*)d_in[6];
  const float* Wk = (const float*)d_in[7];
  const float* bk = (const float*)d_in[8];
  const float* Wv = (const float*)d_in[9];
  const float* bv = (const float*)d_in[10];
  const float* Wp = (const float*)d_in[11];
  const float* ub = (const float*)d_in[12];
  const float* vb = (const float*)d_in[13];
  const float* Wo = (const float*)d_in[14];
  const float* bo = (const float*)d_in[15];
  float* out = (float*)d_out;

  char* ws = (char*)d_ws;
  size_t off = 0;
  auto take = [&](size_t bytes) { char* p = ws + off; off += (bytes + 255) & ~(size_t)255; return p; };

  unsigned short* w_bf   = (unsigned short*)take(5 * 262144 * sizeof(unsigned short));
  unsigned short* wq_bf  = w_bf;
  unsigned short* wk_bf  = w_bf + 1 * 262144;
  unsigned short* wv_bf  = w_bf + 2 * 262144;
  unsigned short* wp_bf  = w_bf + 3 * 262144;
  unsigned short* wo_bf  = w_bf + 4 * 262144;
  float*          zbias  = (float*)take(CC * sizeof(float));
  unsigned short* xnT    = (unsigned short*)take((size_t)BB * TT * CC * 2);          // [N,C]
  unsigned short* posT   = (unsigned short*)take((size_t)PPAD * CC * 2);             // [PPAD,C]
  unsigned short* pT     = (unsigned short*)take((size_t)HH * PPAD * DD * 2);        // [H,PPAD,D]
  unsigned short* qu     = (unsigned short*)take((size_t)BB * HH * TT * DD * 2);     // [B,H,T,D]
  unsigned short* qv     = (unsigned short*)take((size_t)BB * HH * TT * DD * 2);
  unsigned short* kT     = (unsigned short*)take((size_t)BB * HH * TT * DD * 2);
  unsigned short* vbf    = (unsigned short*)take((size_t)BB * HH * DD * TT * 2);     // [B,H,D,T]
  unsigned short* ctxT   = (unsigned short*)take((size_t)BB * TT * CC * 2);          // [N,C]
  float*          rawbuf = (float*)take((size_t)BB * HH * ICHUNK * PPAD * 4);        // 33.5 MB
  float*          score  = (float*)take((size_t)BB * HH * TT * TT * 4);              // 128 MB
  unsigned short* attnT  = (unsigned short*)take((size_t)BB * HH * TT * TT * 2);     // 64 MB

  // 1) weights -> bf16 (+zero bias); pos_emb transpose; layernorm
  convert_w_kernel<<<(5 * 262144 + CC + 255) / 256, 256, 0, stream>>>(Wq, Wk, Wv, Wp, Wo, w_bf, zbias);
  transpose_pos_kernel<<<(PPAD * CC + 255) / 256, 256, 0, stream>>>(pos_emb, posT);
  layernorm_kernel<<<BB * TT, 128, 0, stream>>>(x, ln_g, ln_b, xnT);

  // 2) projections (WMMA GEMMs, K=512); blocks = 4 waves x 16x64 strips
  gemm512_kernel<<<dim3(CC / 64, (BB * TT) / 64), 128, 0, stream>>>(wq_bf, xnT, bq, ub, vb, qu, qv, 0);
  gemm512_kernel<<<dim3(CC / 64, (BB * TT) / 64), 128, 0, stream>>>(wk_bf, xnT, bk, nullptr, nullptr, kT, nullptr, 1);
  gemm512_kernel<<<dim3(CC / 64, (BB * TT) / 64), 128, 0, stream>>>(wv_bf, xnT, bv, nullptr, nullptr, vbf, nullptr, 2);
  gemm512_kernel<<<dim3(CC / 64, PPAD / 64), 128, 0, stream>>>(wp_bf, posT, zbias, nullptr, nullptr, pT, nullptr, 3);

  // 3) content scores into fp32 score buffer
  content_gemm_kernel<<<dim3(TT / 64, TT / 64, BB * HH), 128, 0, stream>>>(qu, kT, score);

  // 4) positional scores: chunked raw GEMM + fused rel_shift add
  for (int c = 0; c < TT / ICHUNK; ++c) {
    pos_gemm_kernel<<<dim3(ICHUNK / 64, PPAD / 64, BB * HH), 128, 0, stream>>>(qv, pT, rawbuf, c * ICHUNK);
    size_t nel = (size_t)BB * HH * ICHUNK * TT;
    shift_add_kernel<<<(unsigned)((nel + 255) / 256), 256, 0, stream>>>(score, rawbuf, c * ICHUNK);
  }

  // 5) scale + mask + softmax -> transposed bf16 attention
  softmax_kernel<<<BB * HH * TT, 256, 0, stream>>>(score, x_mask, attnT);

  // 6) context = V x attn (quirky contraction over query index, per reference)
  context_gemm_kernel<<<dim3(TT / 64, BB * HH), 128, 0, stream>>>(vbf, attnT, ctxT);

  // 7) output projection -> fp32 [B,C,T]
  gemm512_kernel<<<dim3(CC / 64, (BB * TT) / 64), 128, 0, stream>>>(wo_bf, ctxT, bo, nullptr, nullptr, out, nullptr, 4);
}
